// SsdCostVolume_52690658787612
// MI455X (gfx1250) — compile-verified, matrix operand order checked by reading, not run on hardware
//
#include <hip/hip_runtime.h>

// Problem constants (from reference)
#define B_   2
#define C_   16
#define H_   128
#define W_   256
#define CW   48          // CORR_WIDTH
#define R_   3           // H_RADIUS == V_RADIUS
#define WEXT 304         // >= W_ + CW - 1 = 303 (BoxSL support), padded

typedef __attribute__((ext_vector_type(2)))  float    v2f;
typedef __attribute__((ext_vector_type(8)))  float    v8f;
typedef __attribute__((ext_vector_type(16))) _Float16 v16h;

#if defined(__has_builtin)
#  if __has_builtin(__builtin_amdgcn_wmma_f32_16x16x4_f32)
#    define USE_F32_WMMA 1
#  endif
#endif

// ---------------------------------------------------------------------------
// Kernel 1: per-pixel channel sum of squares for both images.
//   SL[b,y,x] = sum_c L^2 ; SR[b,y,x] = sum_c R^2
// ---------------------------------------------------------------------------
__global__ void k_sumsq(const float* __restrict__ L, const float* __restrict__ Rr,
                        float* __restrict__ SL, float* __restrict__ SR) {
    const int NS = B_ * H_ * W_;
    int idx = blockIdx.x * blockDim.x + threadIdx.x;
    if (idx >= 2 * NS) return;
    int which = (idx >= NS) ? 1 : 0;
    int i = which ? idx - NS : idx;
    int b  = i / (H_ * W_);
    int yx = i % (H_ * W_);
    const float* src = (which ? Rr : L) + (size_t)b * C_ * H_ * W_ + yx;
    float s = 0.f;
#pragma unroll
    for (int c = 0; c < C_; ++c) {
        float v = src[(size_t)c * H_ * W_];
        s += v * v;
    }
    (which ? SR : SL)[i] = s;
}

// ---------------------------------------------------------------------------
// Kernel 2: 7x7 zero-padded box filters of SL (on extended width) and SR.
//   BSL[b,y,u] for u in [0,WEXT) ; BSR[b,y,x] for x in [0,W_)
// ---------------------------------------------------------------------------
__global__ void k_boxsums(const float* __restrict__ SL, const float* __restrict__ SR,
                          float* __restrict__ BSL, float* __restrict__ BSR) {
    const int NB = B_ * H_ * WEXT;
    const int NS = B_ * H_ * W_;
    int idx = blockIdx.x * blockDim.x + threadIdx.x;
    if (idx < NB) {
        int u = idx % WEXT; int t = idx / WEXT;
        int y = t % H_;     int b = t / H_;
        float s = 0.f;
        for (int dy = -R_; dy <= R_; ++dy) {
            int yy = y + dy; if ((unsigned)yy >= (unsigned)H_) continue;
            for (int dx = -R_; dx <= R_; ++dx) {
                int xx = u + dx; if ((unsigned)xx >= (unsigned)W_) continue;
                s += SL[((size_t)b * H_ + yy) * W_ + xx];
            }
        }
        BSL[idx] = s;
    } else if (idx < NB + NS) {
        int i = idx - NB;
        int x = i % W_; int t = i / W_;
        int y = t % H_; int b = t / H_;
        float s = 0.f;
        for (int dy = -R_; dy <= R_; ++dy) {
            int yy = y + dy; if ((unsigned)yy >= (unsigned)H_) continue;
            for (int dx = -R_; dx <= R_; ++dx) {
                int xx = x + dx; if ((unsigned)xx >= (unsigned)W_) continue;
                s += SR[((size_t)b * H_ + yy) * W_ + xx];
            }
        }
        BSR[i] = s;
    }
}

// ---------------------------------------------------------------------------
// Kernel 3: banded cross-correlation via WMMA.
// One wave computes one 16x16 tile of Pmat[x, x'] = sum_c R[c,y,x] * L[c,y,x']
// for x in [x0,x0+16), x' in [x1,x1+16). 4 waves/block cover x' band [x0,x0+64).
// Results scattered to P[b,y,x,d] with d = x' - x in [0,48).
// ---------------------------------------------------------------------------
__global__ __launch_bounds__(128) void k_band_gemm(const float* __restrict__ L,
                                                   const float* __restrict__ Rr,
                                                   float* __restrict__ P) {
    const int xt   = blockIdx.x;          // x tile: 0..15
    const int y    = blockIdx.y;          // row:    0..127
    const int b    = blockIdx.z;          // batch:  0..1
    const int lane = threadIdx.x & 31;
    const int jt   = threadIdx.x >> 5;    // band sub-tile: 0..3
    const int x0   = xt * 16;
    const int x1   = x0 + jt * 16;
    const int mn   = lane & 15;           // M (for A) / N (for B) within tile
    const int half = lane >> 4;           // lane-half selects K group

    const float* Lb = L  + (size_t)b * C_ * H_ * W_;
    const float* Rb = Rr + (size_t)b * C_ * H_ * W_;
    const int xc = x1 + mn;               // L column (may exceed W_ -> zero ext)

    v8f acc = {};
#ifdef USE_F32_WMMA
    // V_WMMA_F32_16X16X4_F32: A 16x4 f32 (2 VGPR), B 4x16 f32 (2 VGPR).
    // Lane layout (ISA 7.12.2): lanes 0-15 hold K={0,1}, lanes 16-31 K={2,3}.
    const int kb = half * 2;
#pragma unroll
    for (int c0 = 0; c0 < C_; c0 += 4) {
        int k0 = c0 + kb;
        v2f a, bv;
        a.x = Rb[((size_t)k0       * H_ + y) * W_ + x0 + mn];
        a.y = Rb[((size_t)(k0 + 1) * H_ + y) * W_ + x0 + mn];
        float b0 = 0.f, b1 = 0.f;
        if (xc < W_) {
            b0 = Lb[((size_t)k0       * H_ + y) * W_ + xc];
            b1 = Lb[((size_t)(k0 + 1) * H_ + y) * W_ + xc];
        }
        bv.x = b0; bv.y = b1;
        acc = __builtin_amdgcn_wmma_f32_16x16x4_f32(
            false, a, false, bv, (short)0, acc, false, false);
    }
#else
    // Fallback: V_WMMA_F32_16X16X32_F16 with K=16 channels zero-padded to 32.
    // 16-bit A 16x32 lane layout: element e<8 -> K=e+8*half ; e>=8 -> K=e+8+8*half.
    v16h a, bv;
#pragma unroll
    for (int e = 0; e < 16; ++e) {
        int k = (e < 8) ? (e + 8 * half) : (e + 8 + 8 * half);
        float av = (k < C_) ? Rb[((size_t)k * H_ + y) * W_ + x0 + mn] : 0.f;
        float bb = (k < C_ && xc < W_) ? Lb[((size_t)k * H_ + y) * W_ + xc] : 0.f;
        a[e]  = (_Float16)av;
        bv[e] = (_Float16)bb;
    }
    acc = __builtin_amdgcn_wmma_f32_16x16x32_f16(
        false, a, false, bv, (short)0, acc, false, false);
#endif

    // D layout: VGPR r holds (M = r + 8*half, N = lane&15).
    float* Pb = P + (size_t)(b * H_ + y) * W_ * CW;
    const int hi = half * 8;
#pragma unroll
    for (int r = 0; r < 8; ++r) {
        int m = r + hi;                 // x = x0 + m
        int d = jt * 16 + mn - m;       // disparity
        if (d >= 0 && d < CW)
            Pb[(size_t)(x0 + m) * CW + d] = acc[r];
    }
}

// ---------------------------------------------------------------------------
// Kernel 4: horizontal 7-tap box over x of P -> Q  (per disparity channel)
// ---------------------------------------------------------------------------
__global__ void k_hbox(const float* __restrict__ P, float* __restrict__ Q) {
    const int N = B_ * H_ * W_ * CW;
    int idx = blockIdx.x * blockDim.x + threadIdx.x;
    if (idx >= N) return;
    int d = idx % CW; int t = idx / CW;
    int x = t % W_;   int t2 = t / W_;       // t2 = b*H_ + y
    float s = 0.f;
#pragma unroll
    for (int dx = -R_; dx <= R_; ++dx) {
        int xx = x + dx;
        if ((unsigned)xx < (unsigned)W_) s += P[((size_t)t2 * W_ + xx) * CW + d];
    }
    Q[idx] = s;
}

// ---------------------------------------------------------------------------
// Kernel 5: vertical 7-tap box over y of Q, combine with square terms.
//   cost = BSL[y, x+d] + BSR[y, x] - 2 * Box(P)
// ---------------------------------------------------------------------------
__global__ void k_vbox_final(const float* __restrict__ Q, const float* __restrict__ BSL,
                             const float* __restrict__ BSR, float* __restrict__ out) {
    const int N = B_ * H_ * W_ * CW;
    int idx = blockIdx.x * blockDim.x + threadIdx.x;
    if (idx >= N) return;
    int d = idx % CW; int t  = idx / CW;
    int x = t % W_;   int t2 = t / W_;
    int y = t2 % H_;  int b  = t2 / H_;
    float s = 0.f;
#pragma unroll
    for (int dy = -R_; dy <= R_; ++dy) {
        int yy = y + dy;
        if ((unsigned)yy < (unsigned)H_)
            s += Q[((size_t)(b * H_ + yy) * W_ + x) * CW + d];
    }
    out[idx] = BSL[((size_t)b * H_ + y) * WEXT + x + d]
             + BSR[((size_t)b * H_ + y) * W_ + x]
             - 2.f * s;
}

// ---------------------------------------------------------------------------
extern "C" void kernel_launch(void* const* d_in, const int* in_sizes, int n_in,
                              void* d_out, int out_size, void* d_ws, size_t ws_size,
                              hipStream_t stream) {
    (void)in_sizes; (void)n_in; (void)out_size; (void)ws_size;
    const float* L  = (const float*)d_in[0];   // im_left  [2,16,128,256]
    const float* Rr = (const float*)d_in[1];   // im_right [2,16,128,256]
    float* out = (float*)d_out;                // [2,128,256,48]

    const size_t NP = (size_t)B_ * H_ * W_ * CW;   // 3,145,728
    const size_t NS = (size_t)B_ * H_ * W_;        // 65,536
    const size_t NB = (size_t)B_ * H_ * WEXT;      // 77,824

    float* ws  = (float*)d_ws;
    float* P   = ws;                 // band cross-correlation
    float* Q   = P   + NP;           // horizontally box-filtered P
    float* SL  = Q   + NP;           // sum_c L^2
    float* SR  = SL  + NS;           // sum_c R^2
    float* BSL = SR  + NS;           // 7x7 box of SL on extended width
    float* BSR = BSL + NB;           // 7x7 box of SR

    { // 1: channel sums of squares
        int n = (int)(2 * NS);
        k_sumsq<<<(n + 255) / 256, 256, 0, stream>>>(L, Rr, SL, SR);
    }
    { // 2: 7x7 box of the square terms
        int n = (int)(NB + NS);
        k_boxsums<<<(n + 255) / 256, 256, 0, stream>>>(SL, SR, BSL, BSR);
    }
    { // 3: WMMA banded GEMM -> P[b,y,x,d]
        dim3 grid(W_ / 16, H_, B_);
        k_band_gemm<<<grid, 128, 0, stream>>>(L, Rr, P);
    }
    { // 4: horizontal box of P
        int n = (int)NP;
        k_hbox<<<(n + 255) / 256, 256, 0, stream>>>(P, Q);
    }
    { // 5: vertical box + combine -> output
        int n = (int)NP;
        k_vbox_final<<<(n + 255) / 256, 256, 0, stream>>>(Q, BSL, BSR, out);
    }
}
